// NeuralNetworkOnAEV_31963146616899
// MI455X (gfx1250) — matrix operand order repaired and checked by reading, not compiled, not run on hardware
//
#include <hip/hip_runtime.h>
#include <hip/hip_bf16.h>

typedef __attribute__((ext_vector_type(16))) __bf16 v16bf;
typedef __attribute__((ext_vector_type(8)))  __bf16 v8bf;
typedef __attribute__((ext_vector_type(8)))  float  v8f;

#define CELU_ALPHA 0.1f

__device__ __forceinline__ __bf16 f2bf(float f) {
  union { float f; unsigned u; } v; v.f = f;
  unsigned r = v.u + 0x7FFFu + ((v.u >> 16) & 1u);   // round-to-nearest-even
  union { unsigned short s; __bf16 b; } o; o.s = (unsigned short)(r >> 16);
  return o.b;
}

__device__ __forceinline__ float celu_f(float x) {
  return x > 0.0f ? x : CELU_ALPHA * (__expf(x * (1.0f / CELU_ALPHA)) - 1.0f);
}

union Frag { v16bf v; v8bf h[2]; unsigned short u[16]; };

// One MLP layer on a 32-row activation tile resident in LDS.
// Wt is bf16 [O][I] (pre-transposed, per-species base already applied).
// WMMA bf16 layouts used here: A lane m=l&15, 16 contiguous K per half by l>>4;
// B lane n=l&15, 16 contiguous K per half; C lane n=l&15, row m = r + 8*(l>>4).
template <int I, int O>
__device__ __forceinline__ void mlp_layer(
    const __bf16* __restrict__ bufIn, int strideIn,
    __bf16* __restrict__ bufOut, int strideOut,
    const __bf16* __restrict__ Wt, const float* __restrict__ bias,
    int lane, int wave)
{
  constexpr int ntiles = O >> 4;
  const int khalf = lane >> 4;
  const int nlane = lane & 15;
  for (int t = wave; t < 2 * ntiles; t += 8) {
    const int mtile = t / ntiles;
    const int ntile = t - mtile * ntiles;
    const __bf16* arow = bufIn + (mtile * 16 + nlane) * strideIn;
    const __bf16* brow = Wt + (size_t)(ntile * 16 + nlane) * I + khalf * 16;
    v8f acc = {};
    for (int k0 = 0; k0 < I; k0 += 32) {
      __builtin_prefetch(brow + k0 + 64, 0, 3);   // keep weight panels near (L2-resident)
      Frag a, b;
      a.h[0] = *(const v8bf*)(arow + k0 + khalf * 16);
      a.h[1] = *(const v8bf*)(arow + k0 + khalf * 16 + 8);
      b.h[0] = *(const v8bf*)(brow + k0);
      b.h[1] = *(const v8bf*)(brow + k0 + 8);
      acc = __builtin_amdgcn_wmma_f32_16x16x32_bf16(false, a.v, false, b.v,
                                                    (short)0, acc, false, false);
    }
    const float bv  = bias[ntile * 16 + nlane];
    const int mbase = mtile * 16 + khalf * 8;
    const int col   = ntile * 16 + nlane;
    #pragma unroll
    for (int r = 0; r < 8; ++r)
      bufOut[(mbase + r) * strideOut + col] = f2bf(celu_f(acc[r] + bv));
  }
}

__global__ void __launch_bounds__(256)
ani_mlp_kernel(const float* __restrict__ fullaev, const int* __restrict__ species,
               const __bf16* __restrict__ Wt0, const __bf16* __restrict__ Wt1,
               const __bf16* __restrict__ Wt2, const __bf16* __restrict__ Wt3,
               const float* __restrict__ B0, const float* __restrict__ B1,
               const float* __restrict__ B2, const float* __restrict__ B3,
               float* __restrict__ out, int A)
{
  constexpr int ST0 = 392;   // 384-pad, multiple of 8 (16B aligned v8bf rows)
  constexpr int ST1 = 264;   // 256-pad
  __shared__ alignas(16) float  stage[32 * 384];   // fp32 async-DMA landing zone
  __shared__ alignas(16) __bf16 buf0[32 * ST0];
  __shared__ alignas(16) __bf16 buf1[32 * ST1];

  const int tid  = threadIdx.x;
  const int lane = tid & 31;
  const int wave = tid >> 5;
  const int a    = blockIdx.y;
  const int bb   = blockIdx.x * 32;
  const int s    = species[a];

  // Stage AEV tile [32 x 384] fp32 into LDS via the CDNA5 async memory->LDS
  // path (no VGPR round-trip; tracked by ASYNCcnt). 16 B per lane per issue.
  const size_t rowbase = ((size_t)bb * A + a) * 384;
  {
    const float* gbase = fullaev + rowbase;
    for (int q = tid; q < 32 * 96; q += 256) {          // 96 x 16B per row
      const int m  = q / 96;
      const int j4 = (q - m * 96) * 4;
      const float* gp = gbase + (size_t)m * A * 384 + j4;
      const unsigned ldsa = (unsigned)(uintptr_t)(&stage[m * 384 + j4]);
      asm volatile("global_load_async_to_lds_b128 %0, %1, off"
                   :: "v"(ldsa), "v"((unsigned long long)(uintptr_t)gp)
                   : "memory");
    }
    asm volatile("s_wait_asynccnt 0x0" ::: "memory");
  }
  __syncthreads();

  // LDS->LDS convert pass: fp32 stage -> padded bf16 activation buffer.
  for (int idx = tid; idx < 32 * 384; idx += 256) {
    const int m = idx / 384, i = idx - m * 384;
    buf0[m * ST0 + i] = f2bf(stage[idx]);
  }
  __syncthreads();

  mlp_layer<384, 256>(buf0, ST0, buf1, ST1, Wt0 + (size_t)s * 256 * 384, B0 + s * 256, lane, wave);
  __syncthreads();
  mlp_layer<256, 192>(buf1, ST1, buf0, ST0, Wt1 + (size_t)s * 192 * 256, B1 + s * 192, lane, wave);
  __syncthreads();
  mlp_layer<192, 160>(buf0, ST0, buf1, ST1, Wt2 + (size_t)s * 160 * 192, B2 + s * 160, lane, wave);
  __syncthreads();

  // Final layer 160 -> 1: masked N-tile (only column 0 has weights; EXEC stays
  // all-ones across the WMMA), then sum-pool over atoms via f32 atomics.
  if (wave < 2) {
    const int khalf = lane >> 4, nlane = lane & 15;
    const int mtile = wave;
    const __bf16* arow = buf1 + (mtile * 16 + nlane) * ST1;
    const __bf16* brow = Wt3 + (size_t)s * 160 + khalf * 16;
    v8f acc = {};
    for (int k0 = 0; k0 < 160; k0 += 32) {
      Frag av, bvv;
      av.h[0] = *(const v8bf*)(arow + k0 + khalf * 16);
      av.h[1] = *(const v8bf*)(arow + k0 + khalf * 16 + 8);
      if (nlane == 0) {
        bvv.h[0] = *(const v8bf*)(brow + k0);
        bvv.h[1] = *(const v8bf*)(brow + k0 + 8);
      } else {
        #pragma unroll
        for (int e = 0; e < 16; ++e) bvv.u[e] = 0;
      }
      acc = __builtin_amdgcn_wmma_f32_16x16x32_bf16(false, av.v, false, bvv.v,
                                                    (short)0, acc, false, false);
    }
    if (nlane == 0) {
      const float bv = B3[s];
      const int mbase = bb + mtile * 16 + khalf * 8;
      #pragma unroll
      for (int r = 0; r < 8; ++r)
        atomicAdd(&out[mbase + r], celu_f(acc[r] + bv));
    }
  }
}

// Convert W [S][I][O] fp32 -> Wt [S][O][I] bf16 (contiguous along K for B-frags).
__global__ void wt_transpose_kernel(const float* __restrict__ W, __bf16* __restrict__ Wt,
                                    int I, int O, int total)
{
  for (int idx = blockIdx.x * blockDim.x + threadIdx.x; idx < total;
       idx += gridDim.x * blockDim.x) {
    const int i   = idx % I;
    const int rem = idx / I;
    const int o   = rem % O;
    const int ss  = rem / O;
    Wt[idx] = f2bf(W[((size_t)ss * I + i) * O + o]);
  }
}

__global__ void zero_kernel(float* __restrict__ p, int n) {
  const int i = blockIdx.x * blockDim.x + threadIdx.x;
  if (i < n) p[i] = 0.0f;
}

extern "C" void kernel_launch(void* const* d_in, const int* in_sizes, int n_in,
                              void* d_out, int out_size, void* d_ws, size_t ws_size,
                              hipStream_t stream) {
  const float* fullaev = (const float*)d_in[0];
  const int*   species = (const int*)d_in[1];
  const float* W0 = (const float*)d_in[2]; const float* b0 = (const float*)d_in[3];
  const float* W1 = (const float*)d_in[4]; const float* b1 = (const float*)d_in[5];
  const float* W2 = (const float*)d_in[6]; const float* b2 = (const float*)d_in[7];
  const float* W3 = (const float*)d_in[8]; const float* b3 = (const float*)d_in[9];

  const int A = in_sizes[1];
  const int B = (int)((size_t)in_sizes[0] / ((size_t)A * 384));

  // Workspace: bf16 transposed weights, ~1.4 MB total.
  char* ws = (char*)d_ws;
  __bf16* Wt0 = (__bf16*)(ws);                               // 4*256*384 bf16 = 786432 B
  __bf16* Wt1 = (__bf16*)(ws + 786432);                      // 4*192*256 bf16 = 393216 B
  __bf16* Wt2 = (__bf16*)(ws + 786432 + 393216);             // 4*160*192 bf16 = 245760 B
  __bf16* Wt3 = (__bf16*)(ws + 786432 + 393216 + 245760);    // 4*1*160  bf16 = 1280 B

  wt_transpose_kernel<<<384, 256, 0, stream>>>(W0, Wt0, 384, 256, 4 * 384 * 256);
  wt_transpose_kernel<<<192, 256, 0, stream>>>(W1, Wt1, 256, 192, 4 * 256 * 192);
  wt_transpose_kernel<<<120, 256, 0, stream>>>(W2, Wt2, 192, 160, 4 * 192 * 160);
  wt_transpose_kernel<<<4,   256, 0, stream>>>(W3, Wt3, 160, 1,   4 * 160 * 1);
  zero_kernel<<<(B + 255) / 256, 256, 0, stream>>>((float*)d_out, B);

  dim3 grid(B / 32, A);   // 32 x 256 = 8192 workgroups, 8 wave32 each
  ani_mlp_kernel<<<grid, 256, 0, stream>>>(fullaev, species, Wt0, Wt1, Wt2, Wt3,
                                           b0, b1, b2, b3, (float*)d_out, A);
}
